// BiStackedLSTMOne_36636071035710
// MI455X (gfx1250) — matrix operand order repaired and checked by reading, not compiled, not run on hardware
//
#include <hip/hip_runtime.h>

#define BATCH 2048
#define SEQ   64
#define IN    512
#define HID   256
#define GATES 1024   /* 4*HID */
#define BT    16     /* batch rows per workgroup */

typedef __attribute__((ext_vector_type(16))) __bf16 v16bf;
typedef __attribute__((ext_vector_type(8)))  float  v8f;

union Frag {
    v16bf bf;
    uint4 q[2];
    unsigned short u[16];
};

__device__ __forceinline__ unsigned short f2bf(float x) {
    unsigned int u = __float_as_uint(x);
    unsigned int r = u + 0x7FFFu + ((u >> 16) & 1u);   // round-to-nearest-even
    return (unsigned short)(r >> 16);
}
__device__ __forceinline__ float sigf(float x)   { return 1.0f / (1.0f + __expf(-x)); }
__device__ __forceinline__ float tanh_f(float x) { float e = __expf(-2.0f * x); return (1.0f - e) / (1.0f + e); }

__device__ __forceinline__ v8f splat8(float x) {
    v8f v;
    #pragma unroll
    for (int i = 0; i < 8; ++i) v[i] = x;
    return v;
}

#define WMMA_BF16(A, B, C) \
    __builtin_amdgcn_wmma_f32_16x16x32_bf16(false, (A), false, (B), (short)0, (C), false, false)

// ---------------------------------------------------------------------------
// Pack fp32 weight (4H x K, row-major, gates = output columns) into per-lane
// bf16 B-fragment order: ((nt*KT + kt)*32 + lane)*16 + e, where
// lane holds N = lane%16, h = lane/16: e<8 -> K = kt*32 + 8h + e; e>=8 -> K = kt*32 + 16 + 8h + (e-8).
// ---------------------------------------------------------------------------
__global__ void pack_weight_kernel(const float* __restrict__ W,
                                   unsigned short* __restrict__ out,
                                   int K, int total) {
    int idx = blockIdx.x * blockDim.x + threadIdx.x;
    if (idx >= total) return;
    int e    = idx & 15;
    int lane = (idx >> 4) & 31;
    int tile = idx >> 9;
    int KT   = K >> 5;
    int kt   = tile % KT;
    int nt   = tile / KT;
    int n = nt * 16 + (lane & 15);
    int k = kt * 32 + ((lane >> 4) << 3) + ((e >> 3) << 4) + (e & 7);
    out[idx] = f2bf(W[(size_t)n * K + k]);
}

__global__ void bias_sum_kernel(const float* __restrict__ bih,
                                const float* __restrict__ bhh,
                                float* __restrict__ out) {
    int i = blockIdx.x * blockDim.x + threadIdx.x;
    if (i < GATES) out[i] = bih[i] + bhh[i];
}

// ---- fragment loaders ------------------------------------------------------
// A (16x32 bf16): lane holds M = lane%16; half = lane/16 selects K runs
// [half*8, half*8+8) -> u[0..7] and [16+half*8, ...) -> u[8..15].
__device__ __forceinline__ Frag load_a_lds(const unsigned short* hs, int row, int stride,
                                           int kb, int lh) {
    Frag fr;
    const unsigned short* p = hs + row * stride + kb + lh * 8;
    fr.q[0] = *(const uint4*)(p);
    fr.q[1] = *(const uint4*)(p + 16);
    return fr;
}

__device__ __forceinline__ Frag load_b(const unsigned short* __restrict__ wp,
                                       int nt, int KT, int kt, int lane) {
    Frag fr;
    const uint4* p = (const uint4*)(wp + ((((size_t)nt * KT + kt) * 32 + lane) << 4));
    fr.q[0] = p[0];
    fr.q[1] = p[1];
    return fr;
}

// Pointwise LSTM cell: acc tiles map as acc[gate*2 + blk], wave owns hidden
// cols j in [16w,16w+16) (blk0) and [16w+128,16w+144) (blk1), all four gates.
__device__ __forceinline__ void lstm_update(v8f* acc, v8f* cst, unsigned short* hs,
                                            int wave, int lr, int lh,
                                            float* __restrict__ hout, int b0, bool writeOut) {
    #pragma unroll
    for (int blk = 0; blk < 2; ++blk) {
        v8f gi = acc[blk];
        v8f gf = acc[2 + blk];
        v8f gg = acc[4 + blk];
        v8f go = acc[6 + blk];
        int j0 = 16 * wave + 128 * blk;
        #pragma unroll
        for (int r = 0; r < 8; ++r) {
            float iv = sigf(gi[r]);
            float fv = sigf(gf[r]);
            float gv = tanh_f(gg[r]);
            float ov = sigf(go[r]);
            float cn = fv * cst[blk][r] + iv * gv;
            cst[blk][r] = cn;
            float hv = ov * tanh_f(cn);
            int M = r + 8 * lh;                       // C/D layout: lanes16-31 -> M+8
            hs[M * HID + j0 + lr] = f2bf(hv);
            if (writeOut) hout[(size_t)(b0 + M) * HID + j0 + lr] = hv;
        }
    }
}

// ---------------------------------------------------------------------------
// Persistent 2-layer LSTM stack over T steps for one batch tile of 16 rows.
// time(t) = tbase + tstep*t.  hout gets fp32 h of layer1 at t = T-1.
// ---------------------------------------------------------------------------
__global__ __launch_bounds__(256)
void lstm_stack_kernel(const float* __restrict__ xs,
                       const unsigned short* __restrict__ w0ih,
                       const unsigned short* __restrict__ w0hh,
                       const float* __restrict__ bias0,
                       const unsigned short* __restrict__ w1ih,
                       const unsigned short* __restrict__ w1hh,
                       const float* __restrict__ bias1,
                       float* __restrict__ hout,
                       int T, int tbase, int tstep) {
    __shared__ unsigned short x_s[BT * IN];    // 16 KB  (bf16 x tile, row-major)
    __shared__ unsigned short h0s[BT * HID];   // 8 KB
    __shared__ unsigned short h1s[BT * HID];   // 8 KB

    const int tid  = threadIdx.x;
    const int wave = tid >> 5;
    const int lane = tid & 31;
    const int lr   = lane & 15;
    const int lh   = lane >> 4;
    const int b0   = blockIdx.x * BT;

    for (int i = tid; i < BT * HID; i += 256) { h0s[i] = 0; h1s[i] = 0; }

    float bias0r[8], bias1r[8];
    #pragma unroll
    for (int n = 0; n < 8; ++n) {
        int nt = wave + 8 * n;
        bias0r[n] = bias0[nt * 16 + lr];
        bias1r[n] = bias1[nt * 16 + lr];
    }

    v8f c0st[2], c1st[2];
    c0st[0] = splat8(0.0f); c0st[1] = splat8(0.0f);
    c1st[0] = splat8(0.0f); c1st[1] = splat8(0.0f);

    for (int t = 0; t < T; ++t) {
        const int time = tbase + tstep * t;

        // ---- stage x tile (16 x 512 fp32 -> bf16 in LDS), coalesced ----
        // 16 rows * 64 groups of 8 floats = 1024 groups, 4 per thread.
        #pragma unroll 1
        for (int g = tid; g < BT * (IN / 8); g += 256) {
            int row = g >> 6;                 // IN/8 = 64 groups per row
            int cg  = (g & 63) << 3;
            const float* p = xs + ((size_t)(b0 + row) * SEQ + time) * IN + cg;
            float4 a0 = *(const float4*)(p);
            float4 a1 = *(const float4*)(p + 4);
            union { uint4 q; unsigned short u[8]; } st;
            st.u[0] = f2bf(a0.x); st.u[1] = f2bf(a0.y); st.u[2] = f2bf(a0.z); st.u[3] = f2bf(a0.w);
            st.u[4] = f2bf(a1.x); st.u[5] = f2bf(a1.y); st.u[6] = f2bf(a1.z); st.u[7] = f2bf(a1.w);
            *(uint4*)(x_s + row * IN + cg) = st.q;
            if (t + 1 < T) __builtin_prefetch(p + tstep * IN, 0, 1);
        }
        __syncthreads();

        // ---- layer 0: gates = x @ W0ih^T + h0 @ W0hh^T + b0 ----
        v8f acc[8];
        #pragma unroll
        for (int n = 0; n < 8; ++n) acc[n] = splat8(bias0r[n]);

        #pragma unroll 1
        for (int kt = 0; kt < IN / 32; ++kt) {
            Frag a = load_a_lds(x_s, lr, IN, kt * 32, lh);
            #pragma unroll
            for (int n = 0; n < 8; ++n) {
                Frag b = load_b(w0ih, wave + 8 * n, IN / 32, kt, lane);
                acc[n] = WMMA_BF16(a.bf, b.bf, acc[n]);
            }
        }
        #pragma unroll 1
        for (int kt = 0; kt < HID / 32; ++kt) {
            Frag a = load_a_lds(h0s, lr, HID, kt * 32, lh);
            #pragma unroll
            for (int n = 0; n < 8; ++n) {
                Frag b = load_b(w0hh, wave + 8 * n, HID / 32, kt, lane);
                acc[n] = WMMA_BF16(a.bf, b.bf, acc[n]);
            }
        }
        __syncthreads();                               // all reads of h0s(t-1) done
        lstm_update(acc, c0st, h0s, wave, lr, lh, hout, b0, false);
        __syncthreads();                               // h0s(t) visible

        // ---- layer 1: gates = h0 @ W1ih^T + h1 @ W1hh^T + b1 ----
        #pragma unroll
        for (int n = 0; n < 8; ++n) acc[n] = splat8(bias1r[n]);

        #pragma unroll 1
        for (int kt = 0; kt < HID / 32; ++kt) {
            Frag a = load_a_lds(h0s, lr, HID, kt * 32, lh);
            #pragma unroll
            for (int n = 0; n < 8; ++n) {
                Frag b = load_b(w1ih, wave + 8 * n, HID / 32, kt, lane);
                acc[n] = WMMA_BF16(a.bf, b.bf, acc[n]);
            }
        }
        #pragma unroll 1
        for (int kt = 0; kt < HID / 32; ++kt) {
            Frag a = load_a_lds(h1s, lr, HID, kt * 32, lh);
            #pragma unroll
            for (int n = 0; n < 8; ++n) {
                Frag b = load_b(w1hh, wave + 8 * n, HID / 32, kt, lane);
                acc[n] = WMMA_BF16(a.bf, b.bf, acc[n]);
            }
        }
        __syncthreads();
        lstm_update(acc, c1st, h1s, wave, lr, lh, hout, b0, t == T - 1);
        __syncthreads();
    }
}

// out[b][c] = b3[c] + hF[b] . W3[c][0:256] + hR[b] . W3[c][256:512]
__global__ void head_kernel(const float* __restrict__ hF, const float* __restrict__ hR,
                            const float* __restrict__ W3, const float* __restrict__ b3,
                            float* __restrict__ out) {
    int idx = blockIdx.x * blockDim.x + threadIdx.x;
    if (idx >= BATCH * 10) return;
    int b = idx / 10, c = idx % 10;
    const float* w  = W3 + c * 512;
    const float* hf = hF + (size_t)b * HID;
    const float* hr = hR + (size_t)b * HID;
    float s = b3[c];
    for (int j = 0; j < HID; ++j) s += hf[j] * w[j];
    for (int j = 0; j < HID; ++j) s += hr[j] * w[HID + j];
    out[idx] = s;
}

extern "C" void kernel_launch(void* const* d_in, const int* in_sizes, int n_in,
                              void* d_out, int out_size, void* d_ws, size_t ws_size,
                              hipStream_t stream) {
    const float* xs = (const float*)d_in[0];
    const float* Wih_f0 = (const float*)d_in[1];
    const float* Whh_f0 = (const float*)d_in[2];
    const float* bih_f0 = (const float*)d_in[3];
    const float* bhh_f0 = (const float*)d_in[4];
    const float* Wih_f1 = (const float*)d_in[5];
    const float* Whh_f1 = (const float*)d_in[6];
    const float* bih_f1 = (const float*)d_in[7];
    const float* bhh_f1 = (const float*)d_in[8];
    const float* Wih_r0 = (const float*)d_in[9];
    const float* Whh_r0 = (const float*)d_in[10];
    const float* bih_r0 = (const float*)d_in[11];
    const float* bhh_r0 = (const float*)d_in[12];
    const float* Wih_r1 = (const float*)d_in[13];
    const float* Whh_r1 = (const float*)d_in[14];
    const float* bih_r1 = (const float*)d_in[15];
    const float* bhh_r1 = (const float*)d_in[16];
    const float* W3 = (const float*)d_in[17];
    const float* b3 = (const float*)d_in[18];
    float* out = (float*)d_out;

    // scratch carve-out (256B aligned)
    char* ws = (char*)d_ws;
    size_t off = 0;
    auto carve = [&](size_t bytes) -> void* {
        void* p = ws + off;
        off = (off + bytes + 255) & ~(size_t)255;
        return p;
    };
    unsigned short* p_w0ih_f = (unsigned short*)carve((size_t)GATES * IN  * 2);
    unsigned short* p_w0hh_f = (unsigned short*)carve((size_t)GATES * HID * 2);
    unsigned short* p_w1ih_f = (unsigned short*)carve((size_t)GATES * HID * 2);
    unsigned short* p_w1hh_f = (unsigned short*)carve((size_t)GATES * HID * 2);
    unsigned short* p_w0ih_r = (unsigned short*)carve((size_t)GATES * IN  * 2);
    unsigned short* p_w0hh_r = (unsigned short*)carve((size_t)GATES * HID * 2);
    unsigned short* p_w1ih_r = (unsigned short*)carve((size_t)GATES * HID * 2);
    unsigned short* p_w1hh_r = (unsigned short*)carve((size_t)GATES * HID * 2);
    float* p_bf0 = (float*)carve(GATES * 4);
    float* p_bf1 = (float*)carve(GATES * 4);
    float* p_br0 = (float*)carve(GATES * 4);
    float* p_br1 = (float*)carve(GATES * 4);
    float* p_hF  = (float*)carve((size_t)BATCH * HID * 4);
    float* p_hR  = (float*)carve((size_t)BATCH * HID * 4);
    (void)ws_size; (void)in_sizes; (void)n_in; (void)out_size;

    const int TB = 256;
    const int nBig   = GATES * IN;   // 524288
    const int nSmall = GATES * HID;  // 262144

    pack_weight_kernel<<<(nBig   + TB - 1) / TB, TB, 0, stream>>>(Wih_f0, p_w0ih_f, IN,  nBig);
    pack_weight_kernel<<<(nSmall + TB - 1) / TB, TB, 0, stream>>>(Whh_f0, p_w0hh_f, HID, nSmall);
    pack_weight_kernel<<<(nSmall + TB - 1) / TB, TB, 0, stream>>>(Wih_f1, p_w1ih_f, HID, nSmall);
    pack_weight_kernel<<<(nSmall + TB - 1) / TB, TB, 0, stream>>>(Whh_f1, p_w1hh_f, HID, nSmall);
    pack_weight_kernel<<<(nBig   + TB - 1) / TB, TB, 0, stream>>>(Wih_r0, p_w0ih_r, IN,  nBig);
    pack_weight_kernel<<<(nSmall + TB - 1) / TB, TB, 0, stream>>>(Whh_r0, p_w0hh_r, HID, nSmall);
    pack_weight_kernel<<<(nSmall + TB - 1) / TB, TB, 0, stream>>>(Wih_r1, p_w1ih_r, HID, nSmall);
    pack_weight_kernel<<<(nSmall + TB - 1) / TB, TB, 0, stream>>>(Whh_r1, p_w1hh_r, HID, nSmall);

    bias_sum_kernel<<<(GATES + TB - 1) / TB, TB, 0, stream>>>(bih_f0, bhh_f0, p_bf0);
    bias_sum_kernel<<<(GATES + TB - 1) / TB, TB, 0, stream>>>(bih_f1, bhh_f1, p_bf1);
    bias_sum_kernel<<<(GATES + TB - 1) / TB, TB, 0, stream>>>(bih_r0, bhh_r0, p_br0);
    bias_sum_kernel<<<(GATES + TB - 1) / TB, TB, 0, stream>>>(bih_r1, bhh_r1, p_br1);

    // forward: 32 steps, time = 30 + t ; reverse: 3 steps, time = 63 - t
    lstm_stack_kernel<<<BATCH / BT, 256, 0, stream>>>(
        xs, p_w0ih_f, p_w0hh_f, p_bf0, p_w1ih_f, p_w1hh_f, p_bf1, p_hF, 32, 30, 1);
    lstm_stack_kernel<<<BATCH / BT, 256, 0, stream>>>(
        xs, p_w0ih_r, p_w0hh_r, p_br0, p_w1ih_r, p_w1hh_r, p_br1, p_hR, 3, 63, -1);

    head_kernel<<<(BATCH * 10 + TB - 1) / TB, TB, 0, stream>>>(p_hF, p_hR, W3, b3, out);
}
